// GLMAttention_5617817223387
// MI455X (gfx1250) — matrix operand order repaired and checked by reading, not compiled, not run on hardware
//
#include <hip/hip_runtime.h>
#include <cstdint>

#define BB   4
#define SS   1024
#define HIDD 2048
#define HH   16
#define DD   128

typedef __attribute__((ext_vector_type(16))) __bf16 v16bf;
typedef __attribute__((ext_vector_type(8)))  __bf16 v8bf;
typedef __attribute__((ext_vector_type(8)))  float  v8f;

#define LDS_STRIDE 40   // 32 data + 8 pad elements; 80B row = 20 banks, 16B-aligned

__device__ __forceinline__ unsigned short f2bf(float f) {
  unsigned u = __float_as_uint(f);
  u += 0x7FFFu + ((u >> 16) & 1u);            // round-to-nearest-even
  return (unsigned short)(u >> 16);
}

// ---- async global -> LDS copy, 16B per lane (CDNA5 ch.08, ASYNCcnt-tracked) ----
__device__ __forceinline__ void async_cp16(unsigned lds_addr, const unsigned short* g) {
  asm volatile("global_load_async_to_lds_b128 %0, %1, off"
               :: "v"(lds_addr), "v"((unsigned long long)(uintptr_t)g)
               : "memory");
}
__device__ __forceinline__ void wait_async0() {
  asm volatile("s_wait_asynccnt 0" ::: "memory");
}

// ---- WMMA fragment loaders from GLOBAL (used by scores kernel) ----
// A[M][K] row-major, ISA 16-bit A 16x32 layout: lane L row=mb+(L&15);
// j=0..7 -> K=k0+j+(L>>4)*8 ; j=8..15 -> K=k0+16+(j-8)+(L>>4)*8
__device__ __forceinline__ v16bf load_a_frag(const unsigned short* A, int lda,
                                             int mbase, int k0, int lane) {
  const int row  = mbase + (lane & 15);
  const int half = lane >> 4;
  const unsigned short* p = A + (size_t)row * lda + k0 + half * 8;
  v8bf lo = *(const v8bf*)p;
  v8bf hi = *(const v8bf*)(p + 16);
  v16bf a;
#pragma unroll
  for (int j = 0; j < 8; ++j) { a[j] = lo[j]; a[j + 8] = hi[j]; }
  return a;
}
// Bt[N][K] row-major: lane L col=nb+(L&15); j=0..7 -> K=k0+j+(L>>4)*8 ; j=8..15 -> K=k0+16+(j-8)+(L>>4)*8
__device__ __forceinline__ v16bf load_b_frag(const unsigned short* Bt, int ldb,
                                             int nbase, int k0, int lane) {
  const int row  = nbase + (lane & 15);
  const int half = lane >> 4;
  const unsigned short* p = Bt + (size_t)row * ldb + k0 + half * 8;
  v8bf lo = *(const v8bf*)p;
  v8bf hi = *(const v8bf*)(p + 16);
  v16bf b;
#pragma unroll
  for (int j = 0; j < 8; ++j) { b[j] = lo[j]; b[j + 8] = hi[j]; }
  return b;
}

// ---- WMMA fragment loaders from LDS tiles (row stride LDS_STRIDE) ----
__device__ __forceinline__ v16bf lds_a_frag(const unsigned short* Asb, int mlocal, int lane) {
  const int row  = mlocal + (lane & 15);
  const int half = lane >> 4;
  const unsigned short* p = Asb + row * LDS_STRIDE + half * 8;
  v8bf lo = *(const v8bf*)p;
  v8bf hi = *(const v8bf*)(p + 16);
  v16bf a;
#pragma unroll
  for (int j = 0; j < 8; ++j) { a[j] = lo[j]; a[j + 8] = hi[j]; }
  return a;
}
__device__ __forceinline__ v16bf lds_b_frag(const unsigned short* Bsb, int nlocal, int lane) {
  const int row  = nlocal + (lane & 15);
  const int half = lane >> 4;
  const unsigned short* p = Bsb + row * LDS_STRIDE + half * 8;
  v8bf lo = *(const v8bf*)p;
  v8bf hi = *(const v8bf*)(p + 16);
  v16bf b;
#pragma unroll
  for (int j = 0; j < 8; ++j) { b[j] = lo[j]; b[j + 8] = hi[j]; }
  return b;
}

// ---------------- fp32 -> bf16 conversion ----------------
__global__ void __launch_bounds__(256) cvt_f32_bf16(const float* __restrict__ src,
                                                    unsigned short* __restrict__ dst,
                                                    long long n) {
  long long i = (long long)blockIdx.x * 256 + threadIdx.x;
  if (i < n) dst[i] = f2bf(src[i]);
}

// ---------------- generic bf16 WMMA GEMM with async-LDS double buffering ----------------
// Block: 256 thr = 8 waves as 4x2; each wave -> 32x32 patch => block tile = 128x64.
// Per K-step(32): stage A(128x32)+B(64x32) bf16 via GLOBAL_LOAD_ASYNC_TO_LDS_B128,
// double-buffered; WMMA operands come from ds_load_b128.
__global__ void __launch_bounds__(256) gemm_bf16_wmma(
    const unsigned short* __restrict__ A,
    const unsigned short* __restrict__ Bt,
    const float* __restrict__ bias,         // nullable
    float* __restrict__ Cf,                 // fp32 out (or null)
    unsigned short* __restrict__ Cb,        // bf16 out (or null)
    int M, int N, int K, int ldc,
    long long AzStride, long long BzStride,
    long long CbStride, long long ChStride, int Hmod)
{
  __shared__ unsigned short As[2][128 * LDS_STRIDE];
  __shared__ unsigned short Bs[2][64 * LDS_STRIDE];

  const int t    = threadIdx.x;
  const int lane = t & 31;
  const int w    = t >> 5;
  const int wm   = w & 3, wn = w >> 2;
  const int mb0  = blockIdx.x * 128;
  const int nb0  = blockIdx.y * 64;
  const int z    = blockIdx.z;

  const unsigned short* Az = A  + (size_t)z * AzStride;
  const unsigned short* Bz = Bt + (size_t)z * BzStride;
  const int bz = z / Hmod, hz = z % Hmod;
  const size_t cOff = (size_t)bz * CbStride + (size_t)hz * ChStride;

  const unsigned asBase = (unsigned)(uintptr_t)(&As[0][0]);
  const unsigned bsBase = (unsigned)(uintptr_t)(&Bs[0][0]);
  const unsigned asBufB = 128 * LDS_STRIDE * 2;   // bytes per A buffer
  const unsigned bsBufB = 64  * LDS_STRIDE * 2;   // bytes per B buffer

  // Per-thread staging map: 8-element (16B) chunks, 4 chunks per 32-elem row.
  // A: 128 rows * 4 = 512 chunks -> 2 per thread ; B: 64 rows * 4 = 256 -> 1 per thread.
  const int aRow0 = t >> 1;                 // q=0 chunks 0..255   -> rows 0..127, col (t&1)*... see below
  // use linear chunk ids instead for clarity:
  auto stage = [&](int k0, int buf) {
#pragma unroll
    for (int q = 0; q < 2; ++q) {
      const int c   = q * 256 + t;          // 0..511
      const int row = c >> 2;
      const int col = (c & 3) * 8;
      const unsigned short* g = Az + (size_t)(mb0 + row) * K + k0 + col;
      async_cp16(asBase + buf * asBufB + row * (LDS_STRIDE * 2) + col * 2, g);
    }
    {
      const int row = t >> 2;
      const int col = (t & 3) * 8;
      const unsigned short* g = Bz + (size_t)(nb0 + row) * K + k0 + col;
      async_cp16(bsBase + buf * bsBufB + row * (LDS_STRIDE * 2) + col * 2, g);
    }
  };
  (void)aRow0;

  v8f acc[2][2] = {};

  int buf = 0;
  stage(0, 0);
  wait_async0();
  __syncthreads();

  for (int k0 = 0; k0 < K; k0 += 32) {
    if (k0 + 32 < K) stage(k0 + 32, buf ^ 1);

    v16bf a0 = lds_a_frag(As[buf], wm * 32,      lane);
    v16bf a1 = lds_a_frag(As[buf], wm * 32 + 16, lane);
    v16bf b0 = lds_b_frag(Bs[buf], wn * 32,      lane);
    v16bf b1 = lds_b_frag(Bs[buf], wn * 32 + 16, lane);
    acc[0][0] = __builtin_amdgcn_wmma_f32_16x16x32_bf16(false, a0, false, b0, (short)0, acc[0][0], false, false);
    acc[0][1] = __builtin_amdgcn_wmma_f32_16x16x32_bf16(false, a0, false, b1, (short)0, acc[0][1], false, false);
    acc[1][0] = __builtin_amdgcn_wmma_f32_16x16x32_bf16(false, a1, false, b0, (short)0, acc[1][0], false, false);
    acc[1][1] = __builtin_amdgcn_wmma_f32_16x16x32_bf16(false, a1, false, b1, (short)0, acc[1][1], false, false);

    wait_async0();        // next-buffer async copies landed (per-wave ASYNCcnt)
    __syncthreads();      // all waves done reading buf & all LDS writes visible
    buf ^= 1;
  }

  // C/D layout: lane L -> col = n + (L&15); VGPR r -> row = m + r + 8*(L>>4)
  const int c0 = lane & 15;
  const int rb = (lane >> 4) * 8;
#pragma unroll
  for (int tm = 0; tm < 2; ++tm) {
#pragma unroll
    for (int tn = 0; tn < 2; ++tn) {
      const int col = nb0 + wn * 32 + tn * 16 + c0;
      const float badd = bias ? bias[col] : 0.0f;
#pragma unroll
      for (int r = 0; r < 8; ++r) {
        const int row = mb0 + wm * 32 + tm * 16 + rb + r;
        const size_t idx = cOff + (size_t)row * ldc + col;
        const float v = acc[tm][tn][r] + badd;
        if (Cf) Cf[idx] = v;
        else    Cb[idx] = f2bf(v);
      }
    }
  }
}

// ---------------- RoPE + layout repack ----------------
__global__ void __launch_bounds__(256) rope_pack(
    const float* __restrict__ Qf, const float* __restrict__ Kf,
    const float* __restrict__ Vf, const int* __restrict__ pos_ids,
    unsigned short* __restrict__ Qh, unsigned short* __restrict__ Kh,
    unsigned short* __restrict__ Vt)
{
  const int tid = blockIdx.x * 256 + threadIdx.x;   // B*S*H*32 threads
  const int i = tid & 31;
  const int h = (tid >> 5) & (HH - 1);
  const int s = (tid >> 9) & (SS - 1);
  const int b = tid >> 19;

  const float pid = (float)pos_ids[(b * 2 + 0) * SS + s];
  const float bid = (float)pos_ids[(b * 2 + 1) * SS + s];
  const float invf = __expf(-(float)i * 0.2878231366242557f);  // 10000^(-i/32)
  float s1, c1, s2, c2;
  __sincosf(pid * invf, &s1, &c1);
  __sincosf(bid * invf, &s2, &c2);

  const size_t ib = ((size_t)(b * SS + s)) * HIDD + h * DD;
  const size_t ob = (((size_t)(b * HH + h)) * SS + s) * DD;

  {
    float x0 = Qf[ib + i], x32 = Qf[ib + i + 32], x64 = Qf[ib + i + 64], x96 = Qf[ib + i + 96];
    Qh[ob + i]      = f2bf(x0  * c1 - x32 * s1);
    Qh[ob + i + 32] = f2bf(x32 * c1 + x0  * s1);
    Qh[ob + i + 64] = f2bf(x64 * c2 - x96 * s2);
    Qh[ob + i + 96] = f2bf(x96 * c2 + x64 * s2);
  }
  {
    float x0 = Kf[ib + i], x32 = Kf[ib + i + 32], x64 = Kf[ib + i + 64], x96 = Kf[ib + i + 96];
    Kh[ob + i]      = f2bf(x0  * c1 - x32 * s1);
    Kh[ob + i + 32] = f2bf(x32 * c1 + x0  * s1);
    Kh[ob + i + 64] = f2bf(x64 * c2 - x96 * s2);
    Kh[ob + i + 96] = f2bf(x96 * c2 + x64 * s2);
  }
  {
    const size_t vb = ((size_t)(b * HH + h)) * DD;
#pragma unroll
    for (int q = 0; q < 4; ++q) {
      const int d = i + q * 32;
      Vt[(vb + d) * SS + s] = f2bf(Vf[ib + d]);
    }
  }
}

// ---------------- scores GEMM with fused epilogue (direct global; K=128) ----------------
__global__ void __launch_bounds__(256) attn_scores_wmma(
    const unsigned short* __restrict__ Qh, const unsigned short* __restrict__ Kh,
    const float* __restrict__ pbias, const float* __restrict__ mask,
    const float* __restrict__ prev, float* __restrict__ out)
{
  const int lane = threadIdx.x & 31;
  const int w    = threadIdx.x >> 5;
  const int wm   = w & 3, wn = w >> 2;
  const int mb   = blockIdx.x * 128 + wm * 32;
  const int nb   = blockIdx.y * 64  + wn * 32;
  const int z    = blockIdx.z;                 // b*H + h
  const int b    = z >> 4, h = z & 15;

  const unsigned short* Aq = Qh + (size_t)z * SS * DD;
  const unsigned short* Bk = Kh + (size_t)z * SS * DD;

  v8f acc[2][2] = {};
#pragma unroll
  for (int k0 = 0; k0 < DD; k0 += 32) {
    v16bf a0 = load_a_frag(Aq, DD, mb,      k0, lane);
    v16bf a1 = load_a_frag(Aq, DD, mb + 16, k0, lane);
    v16bf b0 = load_b_frag(Bk, DD, nb,      k0, lane);
    v16bf b1 = load_b_frag(Bk, DD, nb + 16, k0, lane);
    acc[0][0] = __builtin_amdgcn_wmma_f32_16x16x32_bf16(false, a0, false, b0, (short)0, acc[0][0], false, false);
    acc[0][1] = __builtin_amdgcn_wmma_f32_16x16x32_bf16(false, a0, false, b1, (short)0, acc[0][1], false, false);
    acc[1][0] = __builtin_amdgcn_wmma_f32_16x16x32_bf16(false, a1, false, b0, (short)0, acc[1][0], false, false);
    acc[1][1] = __builtin_amdgcn_wmma_f32_16x16x32_bf16(false, a1, false, b1, (short)0, acc[1][1], false, false);
  }

  const float scale = 0.08838834764831845f;    // 1/sqrt(128)
  const int c0 = lane & 15;
  const int rb = (lane >> 4) * 8;
#pragma unroll
  for (int tm = 0; tm < 2; ++tm) {
#pragma unroll
    for (int tn = 0; tn < 2; ++tn) {
      const int tt = nb + tn * 16 + c0;
#pragma unroll
      for (int r = 0; r < 8; ++r) {
        const int ss = mb + tm * 16 + rb + r;
        const size_t ibh = ((size_t)z * SS + ss) * SS + tt;
        const float v = (acc[tm][tn][r] + pbias[((size_t)h * SS + ss) * SS + tt]) * scale
                      + mask[((size_t)b * SS + ss) * SS + tt] + prev[ibh];
        out[ibh] = v;
      }
    }
  }
}

// ---------------- softmax: one wave32 per row of 1024 ----------------
__global__ void __launch_bounds__(256) softmax_rows(const float* __restrict__ scores,
                                                    unsigned short* __restrict__ probs)
{
  const int row  = blockIdx.x * 8 + (threadIdx.x >> 5);
  const int lane = threadIdx.x & 31;
  const float* p = scores + (size_t)row * SS;

  float v[32];
  float m = -3.0e38f;
#pragma unroll
  for (int i = 0; i < 32; ++i) { v[i] = p[lane + i * 32]; m = fmaxf(m, v[i]); }
#pragma unroll
  for (int off = 16; off; off >>= 1) m = fmaxf(m, __shfl_xor(m, off, 32));

  float sum = 0.0f;
#pragma unroll
  for (int i = 0; i < 32; ++i) { v[i] = __expf(v[i] - m); sum += v[i]; }
#pragma unroll
  for (int off = 16; off; off >>= 1) sum += __shfl_xor(sum, off, 32);

  const float inv = 1.0f / sum;
  unsigned short* q = probs + (size_t)row * SS;
#pragma unroll
  for (int i = 0; i < 32; ++i) q[lane + i * 32] = f2bf(v[i] * inv);
}

// ---------------- host-side launcher ----------------
extern "C" void kernel_launch(void* const* d_in, const int* in_sizes, int n_in,
                              void* d_out, int out_size, void* d_ws, size_t ws_size,
                              hipStream_t stream)
{
  const float* query = (const float*)d_in[0];
  const float* key   = (const float*)d_in[1];
  const float* value = (const float*)d_in[2];
  const float* mask  = (const float*)d_in[3];
  const float* pbias = (const float*)d_in[4];
  const float* prev  = (const float*)d_in[5];
  const int*   pids  = (const int*)d_in[6];
  const float* Wq = (const float*)d_in[7];  const float* bq = (const float*)d_in[8];
  const float* Wk = (const float*)d_in[9];  const float* bk = (const float*)d_in[10];
  const float* Wv = (const float*)d_in[11]; const float* bv = (const float*)d_in[12];
  const float* Wo = (const float*)d_in[13]; const float* bo = (const float*)d_in[14];

  float* out_proj = (float*)d_out;                                   // [B,S,HID]
  float* scores   = (float*)d_out + (size_t)BB * SS * HIDD;          // [B,H,S,S]

  char* ws = (char*)d_ws;
  const size_t MB = 1ull << 20;
  unsigned short* Wq_bf = (unsigned short*)(ws +   0 * MB);  // 8MB each
  unsigned short* Wk_bf = (unsigned short*)(ws +   8 * MB);
  unsigned short* Wv_bf = (unsigned short*)(ws +  16 * MB);
  unsigned short* Wo_bf = (unsigned short*)(ws +  24 * MB);
  unsigned short* xq_bf = (unsigned short*)(ws +  32 * MB);  // 16MB each
  unsigned short* xk_bf = (unsigned short*)(ws +  48 * MB);
  unsigned short* xv_bf = (unsigned short*)(ws +  64 * MB);
  float*          Qf    = (float*)         (ws +  80 * MB);  // 32MB each
  float*          Kf    = (float*)         (ws + 112 * MB);
  float*          Vf    = (float*)         (ws + 144 * MB);
  unsigned short* Qh    = (unsigned short*)(ws + 176 * MB);  // 16MB each
  unsigned short* Kh    = (unsigned short*)(ws + 192 * MB);
  unsigned short* Vt    = (unsigned short*)(ws + 208 * MB);
  unsigned short* ctxh  = (unsigned short*)(ws + 224 * MB);  // 16MB
  unsigned short* probs = (unsigned short*)(ws +  32 * MB);  // 128MB, reuses dead x/Qf/Kf/Vf

  const long long nx = (long long)BB * SS * HIDD;   // 8,388,608
  const long long nw = (long long)HIDD * HIDD;      // 4,194,304
  cvt_f32_bf16<<<(unsigned)((nx + 255) / 256), 256, 0, stream>>>(query, xq_bf, nx);
  cvt_f32_bf16<<<(unsigned)((nx + 255) / 256), 256, 0, stream>>>(key,   xk_bf, nx);
  cvt_f32_bf16<<<(unsigned)((nx + 255) / 256), 256, 0, stream>>>(value, xv_bf, nx);
  cvt_f32_bf16<<<(unsigned)((nw + 255) / 256), 256, 0, stream>>>(Wq, Wq_bf, nw);
  cvt_f32_bf16<<<(unsigned)((nw + 255) / 256), 256, 0, stream>>>(Wk, Wk_bf, nw);
  cvt_f32_bf16<<<(unsigned)((nw + 255) / 256), 256, 0, stream>>>(Wv, Wv_bf, nw);
  cvt_f32_bf16<<<(unsigned)((nw + 255) / 256), 256, 0, stream>>>(Wo, Wo_bf, nw);

  dim3 blk(256);
  dim3 gproj(32, 32, 1);   // M=4096/128, N=2048/64
  gemm_bf16_wmma<<<gproj, blk, 0, stream>>>(xq_bf, Wq_bf, bq, Qf, nullptr,
      4096, 2048, 2048, 2048, 0, 0, 0, 0, 1);
  gemm_bf16_wmma<<<gproj, blk, 0, stream>>>(xk_bf, Wk_bf, bk, Kf, nullptr,
      4096, 2048, 2048, 2048, 0, 0, 0, 0, 1);
  gemm_bf16_wmma<<<gproj, blk, 0, stream>>>(xv_bf, Wv_bf, bv, Vf, nullptr,
      4096, 2048, 2048, 2048, 0, 0, 0, 0, 1);

  rope_pack<<<8192, 256, 0, stream>>>(Qf, Kf, Vf, pids, Qh, Kh, Vt);

  attn_scores_wmma<<<dim3(8, 16, 64), blk, 0, stream>>>(Qh, Kh, pbias, mask, prev, scores);

  softmax_rows<<<8192, blk, 0, stream>>>(scores, probs);

  // ctx[b,h,s,d] = probs @ V ; written as bf16 into [B,S,HID] via CbStride/ChStride
  gemm_bf16_wmma<<<dim3(8, 2, 64), blk, 0, stream>>>(probs, Vt, nullptr, nullptr, ctxh,
      1024, 128, 1024, 2048,
      (long long)SS * SS, (long long)DD * SS,
      (long long)SS * HIDD, (long long)DD, HH);

  gemm_bf16_wmma<<<gproj, blk, 0, stream>>>(ctxh, Wo_bf, bo, out_proj, nullptr,
      4096, 2048, 2048, 2048, 0, 0, 0, 0, 1);
}
